// PosteriorDecoder_Qa_79611513799411
// MI455X (gfx1250) — matrix operand order repaired and checked by reading, not compiled, not run on hardware
//
#include <hip/hip_runtime.h>
#include <hip/hip_bf16.h>
#include <math.h>

// ---------------- constants ----------------
#define BB     32
#define TT     256          // TU + TM
#define TM_    128
#define HH     512
#define EE     512
#define VV     32000
#define DBVSL  92           // DBV + SLOT
#define KX     1116         // STATE == GRU_IN
#define KXP    1120         // padded to multiple of 32
#define G3     1536         // 3*H

// workspace layout (floats)
#define WS_SCORE   0                      // 8192   attn scores (atomic)
#define WS_SCM     8192                   // 4096   score_c_m   (atomic)
#define WS_XPAD    12288                  // 32*1120 padded GRU input
#define WS_STPAD   48128                  // 32*1120 padded state
#define WS_WH      83968                  // 32*512  Wh = h@W1.T + b
#define WS_CTX     100352                 // 32*512  context
#define WS_GI      116736                 // 32*1536
#define WS_GH      165888                 // 32*1536
#define WS_ZERO_N  83968                  // zero score..st_pad (covers pad tails)

typedef __attribute__((ext_vector_type(16))) __bf16 v16bf;
typedef __attribute__((ext_vector_type(8)))  float  v8f;

// native hardware convert (lowers to v_cvt_pk_bf16_f32)
__device__ __forceinline__ __bf16 f2bf(float f) { return (__bf16)f; }

// ---- full (unguarded) fragment loaders: hot path, vectorizes to global_load_b128 ----
// A fragment: 16x32 bf16, lane = hi*16+m16; elem j -> K = kk + hi*16 + j
__device__ __forceinline__ v16bf load_afrag(const float* __restrict__ rowp, int kk, int hi) {
  v16bf f;
  const int b0 = kk + hi * 16;
  const int b1 = kk + hi * 16 + 8;
#pragma unroll
  for (int j = 0; j < 8; ++j) f[j]     = f2bf(rowp[b0 + j]);
#pragma unroll
  for (int j = 0; j < 8; ++j) f[8 + j] = f2bf(rowp[b1 + j]);
  return f;
}

// B fragment: 32x16 bf16, lane = hi*16+n16; elem j -> K = kk + hi*16 + j
__device__ __forceinline__ v16bf load_bfrag(const float* __restrict__ rowp, int kk, int hi) {
  v16bf f;
  const int b0 = kk + hi * 16;
#pragma unroll
  for (int j = 0; j < 16; ++j) f[j] = f2bf(rowp[b0 + j]);
  return f;
}

// ---- tail loaders: clamp index (always in-bounds) + select-0; branch-free ----
__device__ __forceinline__ v16bf load_afrag_tail(const float* __restrict__ rowp, int kk, int kmax, int hi) {
  v16bf f;
#pragma unroll
  for (int j = 0; j < 16; ++j) {
    int k = kk + hi * 16 + j;
    int kc = (k < kmax) ? k : (kmax - 1);
    float x = rowp[kc];
    f[j] = f2bf((k < kmax) ? x : 0.f);
  }
  return f;
}

__device__ __forceinline__ v16bf load_bfrag_tail(const float* __restrict__ rowp, int kk, int kmax, int hi) {
  v16bf f;
#pragma unroll
  for (int j = 0; j < 16; ++j) {
    int k = kk + hi * 16 + j;
    int kc = (k < kmax) ? k : (kmax - 1);
    float x = rowp[kc];
    f[j] = f2bf((k < kmax) ? x : 0.f);
  }
  return f;
}

__device__ __forceinline__ v8f wmma_bf16(v16bf a, v16bf b, v8f c) {
  return __builtin_amdgcn_wmma_f32_16x16x32_bf16(false, a, false, b, (short)0, c, false, false);
}

__device__ __forceinline__ float half16_reduce(float v) {
  v += __shfl_xor(v, 1, 32);
  v += __shfl_xor(v, 2, 32);
  v += __shfl_xor(v, 4, 32);
  v += __shfl_xor(v, 8, 32);
  return v;
}

// ---------------- kernels ----------------
__global__ void k_zero(float* p, int n) {
  int i = blockIdx.x * blockDim.x + threadIdx.x;
  if (i < n) p[i] = 0.f;
}

// Wave computes a 32x16 tile: both 16-row M-tiles share one B fragment.
// C[m][n] = sum_k A[m][k]*W[n][k] + bias[n];  M is always 32 here.
__global__ void k_gemm_bias2(const float* __restrict__ A, int lda,
                             const float* __restrict__ Wt, int ldb,
                             const float* __restrict__ bias,
                             float* __restrict__ C, int ldc,
                             int Nt, int kmax) {
  const int lane = threadIdx.x & 31;
  const int hi = lane >> 4, l15 = lane & 15;
  int nt = blockIdx.x * 8 + (threadIdx.x >> 5);
  if (nt >= Nt) return;
  const float* arow0 = A + (size_t)l15 * lda;
  const float* arow1 = A + (size_t)(16 + l15) * lda;
  const float* brow  = Wt + (size_t)(nt * 16 + l15) * ldb;
  v8f acc0 = {}, acc1 = {};
  const int kfull = kmax & ~31;
  for (int kk = 0; kk < kfull; kk += 32) {
    v16bf b = load_bfrag(brow, kk, hi);
    acc0 = wmma_bf16(load_afrag(arow0, kk, hi), b, acc0);
    acc1 = wmma_bf16(load_afrag(arow1, kk, hi), b, acc1);
  }
  if (kfull != kmax) {
    v16bf b = load_bfrag_tail(brow, kfull, kmax, hi);
    acc0 = wmma_bf16(load_afrag_tail(arow0, kfull, kmax, hi), b, acc0);
    acc1 = wmma_bf16(load_afrag_tail(arow1, kfull, kmax, hi), b, acc1);
  }
  int n = nt * 16 + l15;
  float bv = bias[n];
#pragma unroll
  for (int i = 0; i < 8; ++i) {
    int m = hi * 8 + i;
    C[(size_t)m * ldc + n]        = acc0[i] + bv;
    C[(size_t)(16 + m) * ldc + n] = acc1[i] + bv;
  }
}

// attention: score[r] += sum_n tanh(Wh[b][n] + enc[r]·W2[n]) * v[n]
// wave computes a 32-row x 16-col tile (2 WMMAs per B fragment); K = 512 (no tail)
__global__ void k_attn(const float* __restrict__ u, const float* __restrict__ mh,
                       const float* __restrict__ W2 /* attn_W + 512, ld 1024 */,
                       const float* __restrict__ attn_v,
                       const float* __restrict__ Wh, float* __restrict__ score) {
  const int lane = threadIdx.x & 31;
  const int hi = lane >> 4, l15 = lane & 15;
  int tile = blockIdx.x * 8 + (threadIdx.x >> 5);
  const int Nt = 32;                       // 512/16
  int mp = tile / Nt, nt = tile % Nt;      // mp < 256 (pairs of 16-row tiles)
  if (mp >= 256) return;
  int r0 = mp * 32 + l15;
  int r1 = r0 + 16;
  int b0 = r0 >> 8, t0 = r0 & 255;
  int b1 = r1 >> 8, t1 = r1 & 255;
  const float* arow0 = (t0 < TM_) ? (u  + ((size_t)b0 * TM_ + t0) * HH)
                                  : (mh + ((size_t)b0 * TM_ + (t0 - TM_)) * HH);
  const float* arow1 = (t1 < TM_) ? (u  + ((size_t)b1 * TM_ + t1) * HH)
                                  : (mh + ((size_t)b1 * TM_ + (t1 - TM_)) * HH);
  const float* brow = W2 + (size_t)(nt * 16 + l15) * 1024;
  v8f acc0 = {}, acc1 = {};
  for (int kk = 0; kk < HH; kk += 32) {
    v16bf bf = load_bfrag(brow, kk, hi);
    acc0 = wmma_bf16(load_afrag(arow0, kk, hi), bf, acc0);
    acc1 = wmma_bf16(load_afrag(arow1, kk, hi), bf, acc1);
  }
  int n = nt * 16 + l15;
  int bb = (mp * 32) >> 8;                 // batch uniform across the 32-row pair
  float wv  = attn_v[n];
  float whn = Wh[bb * HH + n];
#pragma unroll
  for (int i = 0; i < 8; ++i) {
    float s0 = tanhf(acc0[i] + whn) * wv;
    float s1 = tanhf(acc1[i] + whn) * wv;
    s0 = half16_reduce(s0);
    s1 = half16_reduce(s1);
    if (l15 == 0) {
      atomicAdd(&score[mp * 32 + hi * 8 + i], s0);
      atomicAdd(&score[mp * 32 + 16 + hi * 8 + i], s1);
    }
  }
}

// softmax over T, context = attn_w @ enc, build padded x = [emb | ctx | vec]
__global__ void k_softmax_ctx(const float* __restrict__ score,
                              const float* __restrict__ u, const float* __restrict__ mh,
                              const float* __restrict__ emb, const float* __restrict__ vec,
                              float* __restrict__ ctx, float* __restrict__ x_pad) {
  __shared__ float red[256];
  __shared__ float wsm[256];
  __shared__ float smx, ssum;
  int b = blockIdx.x, tid = threadIdx.x;
  float sv = score[b * TT + tid];
  red[tid] = sv; __syncthreads();
  for (int s = 128; s > 0; s >>= 1) { if (tid < s) red[tid] = fmaxf(red[tid], red[tid + s]); __syncthreads(); }
  if (tid == 0) smx = red[0];
  __syncthreads();
  float e = expf(sv - smx);
  red[tid] = e; __syncthreads();
  for (int s = 128; s > 0; s >>= 1) { if (tid < s) red[tid] += red[tid + s]; __syncthreads(); }
  if (tid == 0) ssum = red[0];
  __syncthreads();
  wsm[tid] = e / ssum;
  __syncthreads();
  for (int c = tid; c < HH; c += 256) {
    float acc = 0.f;
    const float* up = u  + (size_t)b * TM_ * HH + c;
    const float* mp = mh + (size_t)b * TM_ * HH + c;
    for (int t = 0; t < TM_; ++t) acc += wsm[t] * up[(size_t)t * HH];
    for (int t = 0; t < TM_; ++t) acc += wsm[TM_ + t] * mp[(size_t)t * HH];
    ctx[b * HH + c] = acc;
  }
  __syncthreads();
  for (int j = tid; j < KX; j += 256) {
    float v;
    if (j < EE)            v = emb[b * EE + j];
    else if (j < EE + HH)  v = ctx[b * HH + (j - EE)];
    else                   v = vec[b * DBVSL + (j - EE - HH)];
    x_pad[b * KXP + j] = v;
  }
}

// GRU gates; also builds padded st = [h_new | ctx | vec] and writes both hidden outputs
__global__ void k_gru_gates(const float* __restrict__ gi, const float* __restrict__ gh,
                            const float* __restrict__ hprev, const float* __restrict__ ctx,
                            const float* __restrict__ vec,
                            float* __restrict__ st_pad,
                            float* __restrict__ out_hid, float* __restrict__ out_gru) {
  int i = blockIdx.x * blockDim.x + threadIdx.x;
  if (i >= BB * KXP) return;
  int b = i / KXP, c = i % KXP;
  if (c < HH) {
    int j = c;
    float ir = gi[b * G3 + j], iz = gi[b * G3 + HH + j], in_ = gi[b * G3 + 2 * HH + j];
    float hr = gh[b * G3 + j], hz = gh[b * G3 + HH + j], hn_ = gh[b * G3 + 2 * HH + j];
    float r = 1.f / (1.f + expf(-(ir + hr)));
    float z = 1.f / (1.f + expf(-(iz + hz)));
    float n = tanhf(in_ + r * hn_);
    float hp = hprev[b * HH + j];
    float hv = (1.f - z) * n + z * hp;
    st_pad[b * KXP + j] = hv;
    out_hid[b * HH + j] = hv;
    out_gru[b * HH + j] = hv;
  } else if (c < EE + HH) {
    st_pad[b * KXP + c] = ctx[b * HH + (c - HH)];
  } else if (c < KX) {
    st_pad[b * KXP + c] = vec[b * DBVSL + (c - EE - HH)];
  } // c in [1116,1120) stays zero
}

// score_c_m[r] += sum_n tanh(m_hiddens[r]·Wcp[n] + b[n]) * st[b][n]
// wave computes 32-row x 16-col tile (2 WMMAs per B fragment); K = 512 (no tail)
__global__ void k_scorec(const float* __restrict__ mh, const float* __restrict__ Wcp,
                         const float* __restrict__ Wcp_b, const float* __restrict__ st_pad,
                         float* __restrict__ scm) {
  const int lane = threadIdx.x & 31;
  const int hi = lane >> 4, l15 = lane & 15;
  int tile = blockIdx.x * 8 + (threadIdx.x >> 5);
  const int Nt = (KX + 15) / 16;            // 70
  int mp = tile / Nt, nt = tile % Nt;       // mp < 128 (pairs of 16-row tiles)
  if (mp >= 128) return;
  const float* arow0 = mh + (size_t)(mp * 32 + l15) * HH;
  const float* arow1 = mh + (size_t)(mp * 32 + 16 + l15) * HH;
  int nb = nt * 16 + l15;
  int nclamp = (nb < KX) ? nb : (KX - 1);
  const float* brow = Wcp + (size_t)nclamp * HH;
  v8f acc0 = {}, acc1 = {};
  for (int kk = 0; kk < HH; kk += 32) {
    v16bf bf = load_bfrag(brow, kk, hi);
    acc0 = wmma_bf16(load_afrag(arow0, kk, hi), bf, acc0);
    acc1 = wmma_bf16(load_afrag(arow1, kk, hi), bf, acc1);
  }
  bool nvalid = nb < KX;
  int b = (mp * 32) >> 7;                   // batch uniform across the 32-row pair
  float bias = nvalid ? Wcp_b[nb] : 0.f;
  float stv  = nvalid ? st_pad[b * KXP + nb] : 0.f;
#pragma unroll
  for (int i = 0; i < 8; ++i) {
    float s0 = nvalid ? tanhf(acc0[i] + bias) * stv : 0.f;
    float s1 = nvalid ? tanhf(acc1[i] + bias) * stv : 0.f;
    s0 = half16_reduce(s0);
    s1 = half16_reduce(s1);
    if (l15 == 0) {
      atomicAdd(&scm[mp * 32 + hi * 8 + i], s0);
      atomicAdd(&scm[mp * 32 + 16 + hi * 8 + i], s1);
    }
  }
}

// joint softmax over [score_g(32000) | score_c_m(128)], normalize in place, scatter copy-probs
__global__ void k_final(float* __restrict__ out, const float* __restrict__ scm,
                        const int* __restrict__ m_input) {
  __shared__ float red[256];
  __shared__ float smx, sinv;
  int b = blockIdx.x, tid = threadIdx.x;
  float* row = out + (size_t)b * VV;
  float mx = -1e30f;
  for (int v = tid; v < VV; v += 256) mx = fmaxf(mx, row[v]);
  if (tid < TM_) mx = fmaxf(mx, scm[b * TM_ + tid]);
  red[tid] = mx; __syncthreads();
  for (int s = 128; s > 0; s >>= 1) { if (tid < s) red[tid] = fmaxf(red[tid], red[tid + s]); __syncthreads(); }
  if (tid == 0) smx = red[0];
  __syncthreads();
  float sum = 0.f;
  for (int v = tid; v < VV; v += 256) sum += expf(row[v] - smx);
  if (tid < TM_) sum += expf(scm[b * TM_ + tid] - smx);
  red[tid] = sum; __syncthreads();
  for (int s = 128; s > 0; s >>= 1) { if (tid < s) red[tid] += red[tid + s]; __syncthreads(); }
  if (tid == 0) sinv = 1.f / red[0];
  __syncthreads();
  for (int v = tid; v < VV; v += 256) row[v] = expf(row[v] - smx) * sinv;
  __threadfence();
  __syncthreads();
  if (tid < TM_) {
    float pc = expf(scm[b * TM_ + tid] - smx) * sinv;
    atomicAdd(&row[m_input[b * TM_ + tid]], pc);
  }
}

// ---------------- host ----------------
extern "C" void kernel_launch(void* const* d_in, const int* in_sizes, int n_in,
                              void* d_out, int out_size, void* d_ws, size_t ws_size,
                              hipStream_t stream) {
  (void)in_sizes; (void)n_in; (void)out_size; (void)ws_size;
  const float* u_hiddens   = (const float*)d_in[0];
  const int*   m_input     = (const int*)  d_in[1];
  const float* m_hiddens   = (const float*)d_in[3];
  const float* emb_at      = (const float*)d_in[4];
  const float* vec_input   = (const float*)d_in[5];
  const float* last_hidden = (const float*)d_in[6];
  const float* attn_W      = (const float*)d_in[7];
  const float* attn_b      = (const float*)d_in[8];
  const float* attn_v      = (const float*)d_in[9];
  const float* gru_W_ih    = (const float*)d_in[10];
  const float* gru_W_hh    = (const float*)d_in[11];
  const float* gru_b_ih    = (const float*)d_in[12];
  const float* gru_b_hh    = (const float*)d_in[13];
  const float* Wgen_W      = (const float*)d_in[14];
  const float* Wgen_b      = (const float*)d_in[15];
  const float* Wcp_W       = (const float*)d_in[16];
  const float* Wcp_b       = (const float*)d_in[17];

  float* out = (float*)d_out;
  float* ws  = (float*)d_ws;
  float* score  = ws + WS_SCORE;
  float* scm    = ws + WS_SCM;
  float* x_pad  = ws + WS_XPAD;
  float* st_pad = ws + WS_STPAD;
  float* Wh     = ws + WS_WH;
  float* ctx    = ws + WS_CTX;
  float* gi     = ws + WS_GI;
  float* gh     = ws + WS_GH;
  float* out_hid = out + (size_t)BB * VV;            // new_hidden (1,B,H)
  float* out_gru = out_hid + (size_t)BB * HH;        // gru_out    (B,1,H)

  // 1. zero atomic targets + padded tails
  k_zero<<<(WS_ZERO_N + 255) / 256, 256, 0, stream>>>(ws, WS_ZERO_N);
  // 2. Wh = h @ attn_W[:, :H].T + attn_b        (32 n-tiles)
  k_gemm_bias2<<<4, 256, 0, stream>>>(last_hidden, HH, attn_W, 2 * HH, attn_b, Wh, HH, 32, HH);
  // 3. attention scores                          (256 row-pairs x 32 n-tiles)
  k_attn<<<1024, 256, 0, stream>>>(u_hiddens, m_hiddens, attn_W + HH, attn_v, Wh, score);
  // 4. softmax + context + x_pad
  k_softmax_ctx<<<BB, 256, 0, stream>>>(score, u_hiddens, m_hiddens, emb_at, vec_input, ctx, x_pad);
  // 5. gi = x @ W_ih.T + b_ih                    (96 n-tiles, K=1116: 34 full + tail)
  k_gemm_bias2<<<12, 256, 0, stream>>>(x_pad, KXP, gru_W_ih, KX, gru_b_ih, gi, G3, 96, KX);
  // 6. gh = h @ W_hh.T + b_hh                    (96 n-tiles)
  k_gemm_bias2<<<12, 256, 0, stream>>>(last_hidden, HH, gru_W_hh, HH, gru_b_hh, gh, G3, 96, HH);
  // 7. GRU gates + st_pad + hidden outputs
  k_gru_gates<<<(BB * KXP + 255) / 256, 256, 0, stream>>>(gi, gh, last_hidden, ctx, vec_input,
                                                          st_pad, out_hid, out_gru);
  // 8. score_g = st @ Wgen_W.T + b -> d_out      (2000 n-tiles, streams 143MB of Wgen_W)
  k_gemm_bias2<<<250, 256, 0, stream>>>(st_pad, KXP, Wgen_W, KX, Wgen_b, out, VV, 2000, KX);
  // 9. score_c_m fused                           (128 row-pairs x 70 n-tiles)
  k_scorec<<<1120, 256, 0, stream>>>(m_hiddens, Wcp_W, Wcp_b, st_pad, scm);
  // 10. joint softmax in place + scatter
  k_final<<<BB, 256, 0, stream>>>(out, scm, m_input);
}
